// CAM_79611513799033
// MI455X (gfx1250) — matrix operand order repaired and checked by reading, not compiled, not run on hardware
//
#include <hip/hip_runtime.h>

typedef __attribute__((ext_vector_type(16))) _Float16 v16h;
typedef __attribute__((ext_vector_type(8)))  float    v8f;

#define NB      2048
#define ENCN    150
#define CAT     450
#define CATP    480      // padded to 15 K-chunks of 32
#define HID     32
#define KC_N    15
#define SAMP_PB 8        // samples per block in att kernel

__device__ __forceinline__ float fast_tanh(float x) {
#if __has_builtin(__builtin_amdgcn_tanhf)
  return __builtin_amdgcn_tanhf(x);          // v_tanh_f32
#else
  float e = __builtin_amdgcn_exp2f(x * 2.8853900817779268f);
  return 1.0f - 2.0f * __builtin_amdgcn_rcpf(e + 1.0f);
#endif
}

// Sum across the 16 lanes of a wave32 DPP row via rotate-butterfly (no LDS).
__device__ __forceinline__ float rowsum16(float x) {
#if __has_builtin(__builtin_amdgcn_update_dpp)
  x += __int_as_float(__builtin_amdgcn_update_dpp(
      0, __float_as_int(x), 0x128, 0xf, 0xf, true));  // row_ror:8
  x += __int_as_float(__builtin_amdgcn_update_dpp(
      0, __float_as_int(x), 0x124, 0xf, 0xf, true));  // row_ror:4
  x += __int_as_float(__builtin_amdgcn_update_dpp(
      0, __float_as_int(x), 0x122, 0xf, 0xf, true));  // row_ror:2
  x += __int_as_float(__builtin_amdgcn_update_dpp(
      0, __float_as_int(x), 0x121, 0xf, 0xf, true));  // row_ror:1
#else
  x += __shfl_xor(x, 8, 16);
  x += __shfl_xor(x, 4, 16);
  x += __shfl_xor(x, 2, 16);
  x += __shfl_xor(x, 1, 16);
#endif
  return x;
}

// ---------------- Kernel 1: encoders -> feat[b, 450] ----------------
__global__ __launch_bounds__(256)
void enc_kernel(const float* __restrict__ x0, const float* __restrict__ x1,
                const float* __restrict__ x2,
                const float* __restrict__ W0, const float* __restrict__ bi0,
                const float* __restrict__ W1, const float* __restrict__ bi1,
                const float* __restrict__ W2, const float* __restrict__ bi2,
                float* __restrict__ feat) {
  __shared__ float sx[412];
  const int b = blockIdx.x;
  const int tid = threadIdx.x;
  for (int t = tid; t < 409; t += 256) {
    float v;
    if (t < 300)      v = x0[b * 300 + t];
    else if (t < 374) v = x1[b * 74 + (t - 300)];
    else              v = x2[b * 35 + (t - 374)];
    sx[t] = v;
  }
  __syncthreads();
  for (int e = tid; e < CAT; e += 256) {
    const float* W; const float* bb; int dim, xoff, eo;
    if (e < 150)      { W = W0; bb = bi0; dim = 300; xoff = 0;   eo = e; }
    else if (e < 300) { W = W1; bb = bi1; dim = 74;  xoff = 300; eo = e - 150; }
    else              { W = W2; bb = bi2; dim = 35;  xoff = 374; eo = e - 300; }
    float acc = bb[eo];
    const float* wr = W + eo * dim;
    for (int d = 0; d < dim; ++d) acc = fmaf(sx[xoff + d], wr[d], acc);
    feat[b * CAT + e] = acc;
  }
}

// ------- Kernel 2: per-(sample,branch) tanh-outer-product attention GEMM -------
// grid = (NB/SAMP_PB, 3), block = 320 (10 waves, one 16-row M-tile each)
__global__ __launch_bounds__(320)
void att_kernel(const float* __restrict__ feat, const float* __restrict__ affw,
                const float* __restrict__ Ww,   const float* __restrict__ Wc,
                const float* __restrict__ Wh,   float* __restrict__ attfeat) {
  // Wc pre-swizzled into WMMA B-operand layout: [kc][ntile][lane][elem]
  __shared__ alignas(32) _Float16 sWc[KC_N][2][32][16];
  __shared__ alignas(16) float sFeat[CATP];

  const int tid    = threadIdx.x;
  const int lane   = tid & 31;
  const int wave   = tid >> 5;    // 0..9 -> M tile
  const int laneHi = lane >> 4;   // 0 = lanes 0-15, 1 = lanes 16-31
  const int laneN  = lane & 15;
  const int br     = blockIdx.y;
  const float aff  = affw[br];

  // Preload Wc (B operand): element j of lane -> K = kc*32 + 16*(j>>3) + 8*laneHi + (j&7), N = lane&15
  for (int f = tid; f < KC_N * 2 * 32 * 16; f += 320) {
    int j  = f & 15;
    int ln = (f >> 4) & 31;
    int nt = (f >> 9) & 1;
    int kc = f >> 10;
    int h  = nt * 16 + (ln & 15);
    int c  = kc * 32 + ((j & 8) << 1) + ((ln >> 4) << 3) + (j & 7);
    float v = (c < CAT) ? Wc[br * HID * CAT + h * CAT + c] : 0.0f;
    sWc[kc][nt][ln][j] = (_Float16)v;
  }
  // per-lane h-indexed weights (N tile 0: h=laneN, N tile 1: h=16+laneN)
  const float ww0 = Ww[br * HID + laneN];
  const float ww1 = Ww[br * HID + 16 + laneN];
  const float wh0 = Wh[br * HID + laneN];
  const float wh1 = Wh[br * HID + 16 + laneN];
  __syncthreads();

  const int b0s = blockIdx.x * SAMP_PB;
  const int e0  = wave * 16;          // 0..144
  const int kb4 = laneHi * 2;         // float4 offset of this half's K base

  for (int s = 0; s < SAMP_PB; ++s) {
    const int b = b0s + s;
    for (int c = tid; c < CATP; c += 320)
      sFeat[c] = (c < CAT) ? feat[b * CAT + c] : 0.0f;
    __syncthreads();

    const float4* sF4 = (const float4*)sFeat;
    // A row owned by this lane: M = lane&15 (padding rows read junk but are
    // never stored, and contaminate only their own discarded D rows)
    const float encA = sFeat[br * ENCN + e0 + laneN];
    const float sA   = encA * aff;     // att[e,c] = tanh(sA * feat[c])

    v8f acc0 = {};
    v8f acc1 = {};
#pragma unroll
    for (int kc = 0; kc < KC_N; ++kc) {
      // B fragments first so ds latency overlaps the tanh chain
      const v16h bv0 = *(const v16h*)&sWc[kc][0][lane][0];
      const v16h bv1 = *(const v16h*)&sWc[kc][1][lane][0];
      const int q = kc * 8 + kb4;      // c0 + kbase, in float4 units
      const float4 f0 = sF4[q];
      const float4 f1 = sF4[q + 1];
      const float4 f2 = sF4[q + 4];    // c0 + 16 + kbase
      const float4 f3 = sF4[q + 5];
      float t[16] = {f0.x, f0.y, f0.z, f0.w, f1.x, f1.y, f1.z, f1.w,
                     f2.x, f2.y, f2.z, f2.w, f3.x, f3.y, f3.z, f3.w};
      v16h av;
#pragma unroll
      for (int j = 0; j < 16; ++j) av[j] = (_Float16)fast_tanh(sA * t[j]);
      acc0 = __builtin_amdgcn_wmma_f32_16x16x32_f16(
          false, av, false, bv0, (short)0, acc0, false, false);
      acc1 = __builtin_amdgcn_wmma_f32_16x16x32_f16(
          false, av, false, bv1, (short)0, acc1, false, false);
    }

    // Epilogue: D VGPR r -> M = r + 8*laneHi, N = lane&15.
    const int ebase = e0 + laneHi * 8;
    float er[8];
#pragma unroll
    for (int r = 0; r < 8; ++r) er[r] = sFeat[br * ENCN + ebase + r];
#pragma unroll
    for (int r = 0; r < 8; ++r) {
      const int e = ebase + r;
      float h0 = fmaxf(acc0[r] + er[r] * ww0, 0.0f);
      float h1 = fmaxf(acc1[r] + er[r] * ww1, 0.0f);
      float p = rowsum16(h0 * wh0 + h1 * wh1);   // sum over h=0..31
      if (laneN == 0 && e < ENCN)
        attfeat[b * CAT + br * ENCN + e] = p + er[r];
    }
    __syncthreads();
  }
}

// ---------------- Kernel 3: classifier 450 -> 128 -> 7 ----------------
__global__ __launch_bounds__(128)
void cls_kernel(const float* __restrict__ attfeat,
                const float* __restrict__ W1, const float* __restrict__ b1v,
                const float* __restrict__ W2, const float* __restrict__ b2v,
                float* __restrict__ out) {
  __shared__ float sf[CAT];
  __shared__ float sh[128];
  const int b = blockIdx.x;
  const int tid = threadIdx.x;
  for (int c = tid; c < CAT; c += 128) sf[c] = attfeat[b * CAT + c];
  __syncthreads();
  float acc = b1v[tid];
  const float* wr = W1 + tid * CAT;
  for (int c = 0; c < CAT; ++c) acc = fmaf(sf[c], wr[c], acc);
  sh[tid] = acc;
  __syncthreads();
  if (tid < 7) {
    float o = b2v[tid];
    const float* w2r = W2 + tid * 128;
    for (int k = 0; k < 128; ++k) o = fmaf(sh[k], w2r[k], o);
    out[b * 7 + tid] = o;
  }
}

extern "C" void kernel_launch(void* const* d_in, const int* in_sizes, int n_in,
                              void* d_out, int out_size, void* d_ws, size_t ws_size,
                              hipStream_t stream) {
  (void)in_sizes; (void)n_in; (void)out_size; (void)ws_size;
  const float* x0    = (const float*)d_in[0];
  const float* x1    = (const float*)d_in[1];
  const float* x2    = (const float*)d_in[2];
  const float* encW0 = (const float*)d_in[3];
  const float* encb0 = (const float*)d_in[4];
  const float* encW1 = (const float*)d_in[5];
  const float* encb1 = (const float*)d_in[6];
  const float* encW2 = (const float*)d_in[7];
  const float* encb2 = (const float*)d_in[8];
  const float* affw  = (const float*)d_in[9];
  const float* Ww    = (const float*)d_in[10];
  const float* Wc    = (const float*)d_in[11];
  const float* Wh    = (const float*)d_in[12];
  const float* clsW1 = (const float*)d_in[13];
  const float* clsb1 = (const float*)d_in[14];
  const float* clsW2 = (const float*)d_in[15];
  const float* clsb2 = (const float*)d_in[16];

  float* feat    = (float*)d_ws;           // [2048, 450]
  float* attfeat = feat + NB * CAT;        // [2048, 450]

  enc_kernel<<<NB, 256, 0, stream>>>(x0, x1, x2, encW0, encb0, encW1, encb1,
                                     encW2, encb2, feat);
  dim3 g2(NB / SAMP_PB, 3);
  att_kernel<<<g2, 320, 0, stream>>>(feat, affw, Ww, Wc, Wh, attfeat);
  cls_kernel<<<NB, 128, 0, stream>>>(attfeat, clsW1, clsb1, clsW2, clsb2,
                                     (float*)d_out);
}
